// GemmaForCausalLM_43654047596973
// MI455X (gfx1250) — matrix operand order repaired and checked by reading, not compiled, not run on hardware
//
#include <hip/hip_runtime.h>
#include <hip/hip_bf16.h>
#include <math.h>

// ---- model constants (match reference) ----
#define S_LEN   2048
#define D_DIM   2048
#define HD_DIM  256
#define NH_N    8
#define NKV_N   4
#define L_N     4
#define V_DIM   32000
#define FF_DIM  8192
#define WIN     1024
#define NEGBIG  (-2.3819763e38f)

typedef __attribute__((ext_vector_type(16))) __bf16 v16bf;
typedef __attribute__((ext_vector_type(8)))  float  v8f;

__device__ __forceinline__ unsigned short f2bf(float f) {
  unsigned int u = __float_as_uint(f);
  u += 0x7FFFu + ((u >> 16) & 1u);          // round-to-nearest-even
  return (unsigned short)(u >> 16);
}

// =====================================================================
// Tiled bf16 WMMA GEMM:  C[M,N] (+epilogue) = A[M,K] * B[N,K]^T
//  - block tile 128x128, K-step 32, 256 threads = 8 waves (2x4)
//  - each wave: 64x32 tile = 4x2 v_wmma_f32_16x16x32_bf16 accumulators
//  - double-buffered LDS fed by GLOBAL_LOAD_ASYNC_TO_LDS_B128 (ASYNCcnt),
//    pipelined one K-step ahead: issue tile t+1, s_wait_asynccnt 4 for
//    tile t (4 async instrs/wave/tile, in-order completion), barrier,
//    compute 8 WMMAs, barrier.
//  - batched over blockIdx.z; B batch index divided by `bdiv` for GQA.
//  - all dims here are multiples of 128 -> no bounds checks.
// =====================================================================
#define BM 128
#define BN 128
#define BK 32

__global__ __launch_bounds__(256)
void gemm_bf16_wmma(const unsigned short* __restrict__ A,
                    const unsigned short* __restrict__ B,
                    float* __restrict__ C,
                    int K, int lda, int ldb, int ldc,
                    long strideA, long strideB, long strideC,
                    int bdiv, int epi)
{
  __shared__ unsigned short sA[2][BM * BK];
  __shared__ unsigned short sB[2][BN * BK];

  const int z = blockIdx.z;
  A += (long)z * strideA;
  B += (long)(z / bdiv) * strideB;
  C += (long)z * strideC;

  const int m0   = blockIdx.y * BM;
  const int n0   = blockIdx.x * BN;
  const int tid  = threadIdx.x;
  const int lane = tid & 31;
  const int wave = tid >> 5;
  const int wm   = (wave & 1) * 64;   // wave M offset in block tile
  const int wn   = (wave >> 1) * 32;  // wave N offset in block tile

  v8f acc[4][2];
  const v8f vzero = {0.f,0.f,0.f,0.f,0.f,0.f,0.f,0.f};
  for (int i = 0; i < 4; ++i)
    for (int j = 0; j < 2; ++j)
      acc[i][j] = vzero;

  // cooperative tile load mapping: thread t owns 32B of A and 32B of B
  const int row_ld = tid >> 1;          // 0..127
  const int col_ld = (tid & 1) * 16;    // 0 or 16 (shorts)

  const int kg = lane >> 4;             // lane half (K-group select)
  const int lr = lane & 15;

  // issue 4 async global->LDS b128 copies (2 for A tile, 2 for B tile)
  auto issue_tile = [&](int buf, int k0) {
    const unsigned short* gA = A + (long)(m0 + row_ld) * lda + k0 + col_ld;
    const unsigned short* gB = B + (long)(n0 + row_ld) * ldb + k0 + col_ld;
    unsigned lA = (unsigned)(unsigned long long)&sA[buf][row_ld * BK + col_ld];
    unsigned lB = (unsigned)(unsigned long long)&sB[buf][row_ld * BK + col_ld];
    asm volatile("global_load_async_to_lds_b128 %0, %1, off"
                 :: "v"(lA),      "v"((unsigned long long)gA)       : "memory");
    asm volatile("global_load_async_to_lds_b128 %0, %1, off"
                 :: "v"(lA + 16), "v"((unsigned long long)(gA + 8)) : "memory");
    asm volatile("global_load_async_to_lds_b128 %0, %1, off"
                 :: "v"(lB),      "v"((unsigned long long)gB)       : "memory");
    asm volatile("global_load_async_to_lds_b128 %0, %1, off"
                 :: "v"(lB + 16), "v"((unsigned long long)(gB + 8)) : "memory");
  };

  const int nk = K / BK;
  issue_tile(0, 0);                       // prologue: tile 0 in flight

  for (int t = 0; t < nk; ++t) {
    const int cur = t & 1;
    if (t + 1 < nk) {
      issue_tile(cur ^ 1, (t + 1) * BK);  // prefetch next tile
      asm volatile("s_wait_asynccnt 0x4" ::: "memory");  // tile t done
    } else {
      asm volatile("s_wait_asynccnt 0x0" ::: "memory");
    }
    __syncthreads();                      // cross-wave visibility of tile t

    // ---- build fragments, contiguous K order shared by A and B ----
    union Frag { v16bf v; unsigned short u[16]; };
    Frag afr[4], bfr[2];

    // B 32x16 bf16: lane half kg holds K = kg*16 + e, column n = lr
    for (int j = 0; j < 2; ++j) {
      const int n = wn + j * 16 + lr;
      #pragma unroll
      for (int e = 0; e < 16; ++e)
        bfr[j].u[e] = sB[cur][n * BK + kg * 16 + e];
    }
    // A 16x32 bf16: row m = lr, K = kg*16 + e
    for (int i = 0; i < 4; ++i) {
      const int m = wm + i * 16 + lr;
      #pragma unroll
      for (int e = 0; e < 16; ++e)
        afr[i].u[e] = sA[cur][m * BK + kg * 16 + e];
    }

    #pragma unroll
    for (int i = 0; i < 4; ++i)
      #pragma unroll
      for (int j = 0; j < 2; ++j)
        acc[i][j] = __builtin_amdgcn_wmma_f32_16x16x32_bf16(
            false, afr[i].v, false, bfr[j].v,
            (short)0, acc[i][j], false, false);

    __syncthreads();                      // reads done before buffer reuse
  }

  // C/D 16x16 f32 layout: n = lane&15 ; m = vgpr + 8*(lane>>4)
  for (int i = 0; i < 4; ++i)
    for (int j = 0; j < 2; ++j) {
      const int nc = n0 + wn + j * 16 + lr;
      const int mb = m0 + wm + i * 16 + kg * 8;
      #pragma unroll
      for (int v = 0; v < 8; ++v) {
        float c = acc[i][j][v];
        if (epi == 1) c = tanhf(c * (1.0f / 30.0f)) * 30.0f;  // final logit cap
        C[(long)(mb + v) * ldc + nc] = c;
      }
    }
}

// =====================================================================
// elementwise / normalization kernels
// =====================================================================
__global__ void embed_kernel(const int* __restrict__ ids,
                             const float* __restrict__ emb,
                             float* __restrict__ h)
{
  const int s = blockIdx.x;
  const int id = ids[s];
  const float sc = 45.25483399593904f;   // sqrt(2048)
  for (int d = threadIdx.x; d < D_DIM; d += blockDim.x)
    h[(long)s * D_DIM + d] = emb[(long)id * D_DIM + d] * sc;
}

__global__ __launch_bounds__(256)
void rmsnorm_bf16_kernel(const float* __restrict__ x,
                         const float* __restrict__ w,
                         unsigned short* __restrict__ out)
{
  __shared__ float red[256];
  const int s = blockIdx.x;
  const float* xr = x + (long)s * D_DIM;
  float ss = 0.f;
  for (int d = threadIdx.x; d < D_DIM; d += 256) { float v = xr[d]; ss += v * v; }
  red[threadIdx.x] = ss; __syncthreads();
  for (int o = 128; o > 0; o >>= 1) {
    if (threadIdx.x < o) red[threadIdx.x] += red[threadIdx.x + o];
    __syncthreads();
  }
  const float rs = rsqrtf(red[0] * (1.0f / D_DIM) + 1e-6f);
  for (int d = threadIdx.x; d < D_DIM; d += 256)
    out[(long)s * D_DIM + d] = f2bf(xr[d] * rs * (1.0f + w[d]));
}

__global__ __launch_bounds__(256)
void residual_rms_add_kernel(float* __restrict__ h,
                             const float* __restrict__ t,
                             const float* __restrict__ w)
{
  __shared__ float red[256];
  const int s = blockIdx.x;
  const float* tr = t + (long)s * D_DIM;
  float ss = 0.f;
  for (int d = threadIdx.x; d < D_DIM; d += 256) { float v = tr[d]; ss += v * v; }
  red[threadIdx.x] = ss; __syncthreads();
  for (int o = 128; o > 0; o >>= 1) {
    if (threadIdx.x < o) red[threadIdx.x] += red[threadIdx.x + o];
    __syncthreads();
  }
  const float rs = rsqrtf(red[0] * (1.0f / D_DIM) + 1e-6f);
  float* hr = h + (long)s * D_DIM;
  for (int d = threadIdx.x; d < D_DIM; d += 256)
    hr[d] += tr[d] * rs * (1.0f + w[d]);
}

__global__ void f32_to_bf16_kernel(const float* __restrict__ in,
                                   unsigned short* __restrict__ out, long n)
{
  long i = (long)blockIdx.x * blockDim.x + threadIdx.x;
  const long stride = (long)gridDim.x * blockDim.x;
  for (; i < n; i += stride) out[i] = f2bf(in[i]);
}

// RoPE + bf16 convert.  grid: (S, nheads), block: HD/2 = 128 threads
__global__ void rope_bf16_kernel(const float* __restrict__ in,
                                 unsigned short* __restrict__ out, int nheads)
{
  const int s = blockIdx.x, hh = blockIdx.y, i = threadIdx.x;
  const float inv = expf(-((float)i * (1.0f / 128.0f)) * 9.210340371976184f); // theta^-(2i/HD)
  float sn, cs;
  sincosf((float)s * inv, &sn, &cs);
  const long base = ((long)s * nheads + hh) * HD_DIM;
  const float x1 = in[base + i], x2 = in[base + i + 128];
  out[base + i]       = f2bf(x1 * cs - x2 * sn);
  out[base + i + 128] = f2bf(x1 * sn + x2 * cs);
}

// v [S, NKV, HD] f32  ->  vt [NKV, HD, S] bf16  (so PV fits A*B^T form)
__global__ void vT_bf16_kernel(const float* __restrict__ v,
                               unsigned short* __restrict__ vt)
{
  const int idx = blockIdx.x;            // kv*HD + d
  const int kv = idx >> 8, d = idx & 255;
  for (int s = threadIdx.x; s < S_LEN; s += blockDim.x)
    vt[(long)idx * S_LEN + s] = f2bf(v[((long)s * NKV_N + kv) * HD_DIM + d]);
}

// scale -> softcap(tanh*50) -> mask -> softmax ; writes bf16 probs.
// grid: (S rows, NH heads), 256 threads, 8 elems/thread (S=2048)
__global__ __launch_bounds__(256)
void softmax_cap_kernel(const float* __restrict__ scores,
                        unsigned short* __restrict__ probs, int is_local)
{
  __shared__ float red[256];
  const int i = blockIdx.x, hh = blockIdx.y;
  const float* row = scores + ((long)hh * S_LEN + i) * S_LEN;
  unsigned short* prow = probs + ((long)hh * S_LEN + i) * S_LEN;

  float vals[8];
  float mx = -3.4e38f;
  for (int t = 0; t < 8; ++t) {
    const int j = threadIdx.x + t * 256;
    float q = row[j] * 0.0625f;                   // HD^-0.5
    q = tanhf(q * (1.0f / 50.0f)) * 50.0f;        // softcap
    const bool masked = (j > i) || (is_local && (i - j >= WIN));
    q = masked ? NEGBIG : q;
    vals[t] = q;
    mx = fmaxf(mx, q);
  }
  red[threadIdx.x] = mx; __syncthreads();
  for (int o = 128; o > 0; o >>= 1) {
    if (threadIdx.x < o) red[threadIdx.x] = fmaxf(red[threadIdx.x], red[threadIdx.x + o]);
    __syncthreads();
  }
  mx = red[0]; __syncthreads();

  float sum = 0.f;
  for (int t = 0; t < 8; ++t) { vals[t] = __expf(vals[t] - mx); sum += vals[t]; }
  red[threadIdx.x] = sum; __syncthreads();
  for (int o = 128; o > 0; o >>= 1) {
    if (threadIdx.x < o) red[threadIdx.x] += red[threadIdx.x + o];
    __syncthreads();
  }
  const float inv = 1.0f / red[0];
  for (int t = 0; t < 8; ++t)
    prow[threadIdx.x + t * 256] = f2bf(vals[t] * inv);
}

// gelu(tanh approx)(gate) * up  -> bf16
__global__ void geluglu_kernel(const float* __restrict__ gate,
                               const float* __restrict__ up,
                               unsigned short* __restrict__ out, long n)
{
  long i = (long)blockIdx.x * blockDim.x + threadIdx.x;
  const long stride = (long)gridDim.x * blockDim.x;
  for (; i < n; i += stride) {
    const float x = gate[i];
    const float g = 0.5f * x * (1.0f + tanhf(0.7978845608028654f * (x + 0.044715f * x * x * x)));
    out[i] = f2bf(g * up[i]);
  }
}

// =====================================================================
// host orchestration
// =====================================================================
extern "C" void kernel_launch(void* const* d_in, const int* in_sizes, int n_in,
                              void* d_out, int out_size, void* d_ws, size_t ws_size,
                              hipStream_t stream) {
  (void)in_sizes; (void)n_in; (void)out_size; (void)ws_size;

  const int*   ids    = (const int*)  d_in[0];
  const float* emb    = (const float*)d_in[1];
  const float* wq     = (const float*)d_in[2];
  const float* wk     = (const float*)d_in[3];
  const float* wv     = (const float*)d_in[4];
  const float* wo     = (const float*)d_in[5];
  const float* wg     = (const float*)d_in[6];
  const float* wu     = (const float*)d_in[7];
  const float* wd     = (const float*)d_in[8];
  const float* ln_in  = (const float*)d_in[9];
  const float* ln_pa  = (const float*)d_in[10];
  const float* ln_pf  = (const float*)d_in[11];
  const float* ln_pff = (const float*)d_in[12];
  const float* ln_f   = (const float*)d_in[13];
  const float* w_lm   = (const float*)d_in[14];

  // ---- workspace carve-up ----
  size_t off = 0;
  auto carve = [&](size_t bytes) -> void* {
    void* p = (char*)d_ws + off;
    off += (bytes + 255) & ~(size_t)255;
    return p;
  };
  float*          h      = (float*)carve((size_t)S_LEN * D_DIM * 4);
  unsigned short* xb     = (unsigned short*)carve((size_t)S_LEN * D_DIM * 2);
  unsigned short* wbuf   = (unsigned short*)carve((size_t)V_DIM * D_DIM * 2);  // reused per-GEMM
  float*          q      = (float*)carve((size_t)S_LEN * NH_N  * HD_DIM * 4);
  float*          k      = (float*)carve((size_t)S_LEN * NKV_N * HD_DIM * 4);
  float*          v      = (float*)carve((size_t)S_LEN * NKV_N * HD_DIM * 4);
  unsigned short* qb     = (unsigned short*)carve((size_t)S_LEN * NH_N  * HD_DIM * 2);
  unsigned short* kb     = (unsigned short*)carve((size_t)S_LEN * NKV_N * HD_DIM * 2);
  unsigned short* vt     = (unsigned short*)carve((size_t)NKV_N * HD_DIM * S_LEN * 2);
  float*          scores = (float*)carve((size_t)NH_N * S_LEN * S_LEN * 4);
  unsigned short* probs  = (unsigned short*)carve((size_t)NH_N * S_LEN * S_LEN * 2);
  float*          attn   = (float*)carve((size_t)S_LEN * NH_N * HD_DIM * 4);
  unsigned short* ab     = (unsigned short*)carve((size_t)S_LEN * NH_N * HD_DIM * 2);
  float*          tmp    = (float*)carve((size_t)S_LEN * D_DIM * 4);
  float*          gate   = (float*)carve((size_t)S_LEN * FF_DIM * 4);
  float*          up     = (float*)carve((size_t)S_LEN * FF_DIM * 4);
  unsigned short* gub    = (unsigned short*)carve((size_t)S_LEN * FF_DIM * 2);

  auto conv = [&](const float* in, unsigned short* outp, long n) {
    int blocks = (int)((n + 2047) / 2048);
    f32_to_bf16_kernel<<<blocks, 256, 0, stream>>>(in, outp, n);
  };
  auto gemm = [&](const unsigned short* A, const unsigned short* Bw, float* Cc,
                  int K, int lda, int ldb, int ldc, int nbn, int batches,
                  long sA, long sB, long sC, int bdiv, int epi) {
    dim3 g(nbn, S_LEN / BM, batches);
    gemm_bf16_wmma<<<g, 256, 0, stream>>>(A, Bw, Cc, K, lda, ldb, ldc,
                                          sA, sB, sC, bdiv, epi);
  };

  embed_kernel<<<S_LEN, 256, 0, stream>>>(ids, emb, h);

  const long wq_ls = (long)NH_N * HD_DIM * D_DIM;   // 2048*2048
  const long wk_ls = (long)NKV_N * HD_DIM * D_DIM;  // 1024*2048
  const long wo_ls = (long)D_DIM * NH_N * HD_DIM;
  const long wg_ls = (long)FF_DIM * D_DIM;
  const long wd_ls = (long)D_DIM * FF_DIM;

  for (int l = 0; l < L_N; ++l) {
    const int is_local = (l % 2 == 0) ? 1 : 0;   // ('local','global','local','global')

    // --- attention ---
    rmsnorm_bf16_kernel<<<S_LEN, 256, 0, stream>>>(h, ln_in + (long)l * D_DIM, xb);

    conv(wq + l * wq_ls, wbuf, wq_ls);
    gemm(xb, wbuf, q, D_DIM, D_DIM, D_DIM, NH_N * HD_DIM, 16, 1, 0, 0, 0, 1, 0);
    conv(wk + l * wk_ls, wbuf, wk_ls);
    gemm(xb, wbuf, k, D_DIM, D_DIM, D_DIM, NKV_N * HD_DIM, 8, 1, 0, 0, 0, 1, 0);
    conv(wv + l * wk_ls, wbuf, wk_ls);
    gemm(xb, wbuf, v, D_DIM, D_DIM, D_DIM, NKV_N * HD_DIM, 8, 1, 0, 0, 0, 1, 0);

    rope_bf16_kernel<<<dim3(S_LEN, NH_N), 128, 0, stream>>>(q, qb, NH_N);
    rope_bf16_kernel<<<dim3(S_LEN, NKV_N), 128, 0, stream>>>(k, kb, NKV_N);
    vT_bf16_kernel<<<NKV_N * HD_DIM, 256, 0, stream>>>(v, vt);

    // scores[h] = Q_h (S x HD) * K_{h/2}^T    K-dim = HD
    gemm(qb, kb, scores, HD_DIM, NH_N * HD_DIM, NKV_N * HD_DIM, S_LEN,
         S_LEN / BN, NH_N, (long)HD_DIM, (long)HD_DIM, (long)S_LEN * S_LEN, 2, 0);

    softmax_cap_kernel<<<dim3(S_LEN, NH_N), 256, 0, stream>>>(scores, probs, is_local);

    // attn[:, h*HD:] = P_h (S x S) * Vt_{h/2}^T   K-dim = S
    gemm(probs, vt, attn, S_LEN, S_LEN, S_LEN, NH_N * HD_DIM,
         HD_DIM / BN, NH_N, (long)S_LEN * S_LEN, (long)HD_DIM * S_LEN, (long)HD_DIM, 2, 0);

    conv(attn, ab, (long)S_LEN * NH_N * HD_DIM);
    conv(wo + l * wo_ls, wbuf, wo_ls);
    gemm(ab, wbuf, tmp, NH_N * HD_DIM, NH_N * HD_DIM, NH_N * HD_DIM, D_DIM, 16, 1, 0, 0, 0, 1, 0);

    residual_rms_add_kernel<<<S_LEN, 256, 0, stream>>>(h, tmp, ln_pa + (long)l * D_DIM);

    // --- MLP ---
    rmsnorm_bf16_kernel<<<S_LEN, 256, 0, stream>>>(h, ln_pf + (long)l * D_DIM, xb);

    conv(wg + l * wg_ls, wbuf, wg_ls);
    gemm(xb, wbuf, gate, D_DIM, D_DIM, D_DIM, FF_DIM, FF_DIM / BN, 1, 0, 0, 0, 1, 0);
    conv(wu + l * wg_ls, wbuf, wg_ls);
    gemm(xb, wbuf, up, D_DIM, D_DIM, D_DIM, FF_DIM, FF_DIM / BN, 1, 0, 0, 0, 1, 0);

    geluglu_kernel<<<(int)(((long)S_LEN * FF_DIM + 2047) / 2048), 256, 0, stream>>>(
        gate, up, gub, (long)S_LEN * FF_DIM);

    conv(wd + l * wd_ls, wbuf, wd_ls);
    gemm(gub, wbuf, tmp, FF_DIM, FF_DIM, FF_DIM, D_DIM, 16, 1, 0, 0, 0, 1, 0);

    residual_rms_add_kernel<<<S_LEN, 256, 0, stream>>>(h, tmp, ln_pff + (long)l * D_DIM);
  }

  // --- final norm + LM head with fused tanh cap, written straight to d_out ---
  rmsnorm_bf16_kernel<<<S_LEN, 256, 0, stream>>>(h, ln_f, xb);
  conv(w_lm, wbuf, (long)V_DIM * D_DIM);
  gemm(xb, wbuf, (float*)d_out, D_DIM, D_DIM, D_DIM, V_DIM, V_DIM / BN, 1, 0, 0, 0, 1, 1);
}
